// PILayer_39539468927515
// MI455X (gfx1250) — matrix operand order repaired and checked by reading, not compiled, not run on hardware
//
#include <hip/hip_runtime.h>

// ---- problem constants (from reference) ----
#define N_PROP   64
#define N_OUT    64
#define N_BASIS  10
#define KDIM     128            // 2*N_PROP
#define NCOL     640            // N_OUT*N_BASIS
#define TILE_M   16
#define WAVES    8
#define BLOCK    (WAVES * 32)

// ---- LDS layout (dynamic shared) ----
#define WT_STRIDE   136                         // 128 + 8 halves pad (272B rows -> no bank clash)
#define LDS_W_BYTES (NCOL * WT_STRIDE * 2)      // 174080: W^T as f16, [n][k]
#define LDS_BIAS_OFF  LDS_W_BYTES
#define LDS_BIAS_BYTES (NCOL * 4)               // 2560
#define A_STRIDE     136                        // padded like W
#define PW_A_BYTES   (TILE_M * A_STRIDE * 2)    // 4352: staged f16 A tile
#define PW_OUT_BYTES (TILE_M * N_OUT * 4)       // 4096: f32 output accumulator tile
#define PW_BAS_BYTES (TILE_M * N_BASIS * 4)     // 640:  transposed basis tile [10][16]
#define PW_BYTES     (PW_A_BYTES + PW_OUT_BYTES + PW_BAS_BYTES)
#define LDS_WAVE_OFF (LDS_W_BYTES + LDS_BIAS_BYTES)
#define LDS_TOTAL    (LDS_WAVE_OFF + WAVES * PW_BYTES)   // 249,344 B  (<= 320KB WGP LDS)

typedef _Float16 half_t;
typedef __attribute__((ext_vector_type(16))) _Float16 v16h;
typedef __attribute__((ext_vector_type(8)))  _Float16 v8h;
typedef __attribute__((ext_vector_type(8)))  float    v8f;

// __builtin_amdgcn_cvt_pkrtz returns an __fp16 vector (distinct type from
// _Float16 in clang) -- keep the packing pipeline in __fp16 vectors and
// store through a pointer cast (bit-identical layout).
typedef __attribute__((ext_vector_type(2)))  __fp16 v2hp;
typedef __attribute__((ext_vector_type(4)))  __fp16 v4hp;
typedef __attribute__((ext_vector_type(8)))  __fp16 v8hp;

static __device__ __forceinline__ v16h join8(v8h lo, v8h hi) {
    return __builtin_shufflevector(lo, hi, 0,1,2,3,4,5,6,7,8,9,10,11,12,13,14,15);
}

// Branchless tanh: CDNA5's V_TANH_F32 trans op (confirmed lowering in round 3);
// fallback is 1 - 2/(exp(2x)+1) via v_exp_f32 + v_rcp_f32, which saturates
// correctly at +/-1 through inf/0 (no EXEC-splitting branches).
static __device__ __forceinline__ float fast_tanh(float x) {
#if __has_builtin(__builtin_amdgcn_tanhf)
    return __builtin_amdgcn_tanhf(x);
#else
    const float e = __builtin_amdgcn_exp2f(x * 2.8853900817779268f); // 2*log2(e)
    return 1.0f - 2.0f * __builtin_amdgcn_rcpf(e + 1.0f);
#endif
}

__global__ __launch_bounds__(BLOCK) void pilayer_wmma_kernel(
    const int*   __restrict__ ind2,    // (nPairs, 2)
    const float* __restrict__ prop,    // (N_ATOMS, 64)  -- L2-resident (12.8MB)
    const float* __restrict__ basis,   // (nPairs, 10)
    const float* __restrict__ W,       // (128, 640)
    const float* __restrict__ bias,    // (640,)
    float*       __restrict__ out,     // (nPairs, 64)
    int nPairs, int nTiles)
{
    extern __shared__ char smem[];
    half_t* wT    = (half_t*)smem;                       // [NCOL][WT_STRIDE] f16
    float*  biasS = (float*)(smem + LDS_BIAS_OFF);       // [NCOL] f32

    const int tid  = threadIdx.x;
    const int lane = tid & 31;
    const int wave = tid >> 5;

    // ---------- Phase 1: stage W^T (f32 -> f16, packed cvt) and bias into LDS ----------
    for (int idx = tid; idx < NCOL * (KDIM / 2); idx += BLOCK) {
        const int kp = idx / NCOL;               // k pair index: k = 2*kp, 2*kp+1
        const int n  = idx - kp * NCOL;          // consecutive tid -> consecutive n (coalesced)
        const float a0 = W[(2 * kp)     * NCOL + n];
        const float a1 = W[(2 * kp + 1) * NCOL + n];
        *(v2hp*)(wT + n * WT_STRIDE + 2 * kp) =
            __builtin_amdgcn_cvt_pkrtz(a0, a1);  // v_cvt_pk_rtz_f16_f32 + ds_store_b32
    }
    for (int n = tid; n < NCOL; n += BLOCK) biasS[n] = bias[n];
    __syncthreads();

    // per-wave private LDS regions (no cross-wave sharing -> no barriers needed below)
    char*   wbase = smem + LDS_WAVE_OFF + wave * PW_BYTES;
    half_t* aS    = (half_t*)wbase;                              // [16][A_STRIDE] f16
    float*  outW  = (float*)(wbase + PW_A_BYTES);                // [16][64] f32
    float*  basWT = (float*)(wbase + PW_A_BYTES + PW_OUT_BYTES); // [10][16] f32 (transposed!)

    const int row = lane & 15;   // A: M row / B: N column owned by this lane
    const int kg  = lane >> 4;   // K-group selector (ISA 16-bit A/B VGPR layout)

    // ---------- Phase 2: each wave streams 16-pair M-tiles ----------
    for (int tile = blockIdx.x * WAVES + wave; tile < nTiles; tile += gridDim.x * WAVES) {
        const int pair0 = tile * TILE_M;

        // prefetch next tile's index rows (global_prefetch_b8)
        {
            const int nxt = tile + gridDim.x * WAVES;
            if (nxt < nTiles)
                __builtin_prefetch(ind2 + (size_t)nxt * TILE_M * 2, 0, 1);
        }

        // --- gather: each lane owns one (pair,atom-half) row of 64 props, cvt f32->f16 ---
        {
            int pg = pair0 + row; if (pg >= nPairs) pg = nPairs - 1;
            const int atom = ind2[pg * 2 + kg];              // kg: 0 -> i atom, 1 -> j atom
            const float4* src = (const float4*)(prop + (size_t)atom * N_PROP);
            half_t* dst = aS + row * A_STRIDE + kg * N_PROP;
            #pragma unroll
            for (int c = 0; c < 8; ++c) {                    // 16 x global_load_b128 total
                const float4 v0 = src[2 * c], v1 = src[2 * c + 1];
                const v2hp p0 = __builtin_amdgcn_cvt_pkrtz(v0.x, v0.y);
                const v2hp p1 = __builtin_amdgcn_cvt_pkrtz(v0.z, v0.w);
                const v2hp p2 = __builtin_amdgcn_cvt_pkrtz(v1.x, v1.y);
                const v2hp p3 = __builtin_amdgcn_cvt_pkrtz(v1.z, v1.w);
                const v4hp q0 = __builtin_shufflevector(p0, p1, 0,1,2,3);
                const v4hp q1 = __builtin_shufflevector(p2, p3, 0,1,2,3);
                *(v8hp*)(dst + 8 * c) =
                    __builtin_shufflevector(q0, q1, 0,1,2,3,4,5,6,7);  // ds_store_b128
            }
        }
        // --- stage the basis tile TRANSPOSED: basWT[b][p] = basis[pair0+p][b] ---
        // (the 8 weights a lane needs per N-tile become 32 contiguous bytes)
        for (int i = lane; i < TILE_M * N_BASIS; i += 32) {
            const int b = i >> 4;            // 0..9
            const int p = i & 15;            // 0..15
            int pg = pair0 + p; if (pg >= nPairs) pg = nPairs - 1;
            basWT[i] = basis[(size_t)pg * N_BASIS + b];
        }
        // --- zero the output accumulator tile ---
        #pragma unroll
        for (int i = 0; i < (TILE_M * N_OUT) / 32; ++i)
            outW[lane + 32 * i] = 0.0f;

        // --- build the four K=32 A fragments (ISA 7.12.2 16-bit A layout) ---
        v16h afrag[4];
        #pragma unroll
        for (int t = 0; t < 4; ++t) {
            const half_t* ap = aS + row * A_STRIDE + t * 32;
            const v8h lo = *(const v8h*)(ap + kg * 8);        // VGPR0-3: K = kg*8 + 0..7
            const v8h hi = *(const v8h*)(ap + 16 + kg * 8);   // VGPR4-7: K = 16 + kg*8 + 0..7
            afrag[t] = join8(lo, hi);
        }

        // col = nt*16 + row; carry bidx = col % 10 and ocol = col / 10 incrementally
        // (col advances by 16 => bidx += 6 (mod 10), ocol += 1 or 2 on wrap)
        int bidx = (row < 10) ? row : row - 10;
        int ocol = (row < 10) ? 0 : 1;

        // --- sweep the 40 N-tiles: 4 chained v_wmma_f32_16x16x32_f16 each ---
        #pragma unroll 2
        for (int nt = 0; nt < NCOL / 16; ++nt) {
            v8f acc = {};
            #pragma unroll
            for (int t = 0; t < 4; ++t) {
                // B 32x16: lane owns column (nt*16+row), K = kg*8 + 0..7 and 16 + kg*8 + 0..7
                const half_t* bp = wT + (nt * 16 + row) * WT_STRIDE + t * 32 + kg * 8;
                const v8h lo = *(const v8h*)bp;
                const v8h hi = *(const v8h*)(bp + 16);
                acc = __builtin_amdgcn_wmma_f32_16x16x32_f16(
                        false, afrag[t], false, join8(lo, hi),
                        (short)0, acc, false, false);
            }
            // --- epilogue: issue ALL LDS loads first (bias + two b128 weight loads),
            //     then 8x tanh/mul/ds_add with no interleaved loads/waits ---
            const float  bv = biasS[nt * 16 + row];
            const float4 wA = *(const float4*)(basWT + bidx * 16 + kg * 8);
            const float4 wB = *(const float4*)(basWT + bidx * 16 + kg * 8 + 4);
            const float wgt[8] = { wA.x, wA.y, wA.z, wA.w, wB.x, wB.y, wB.z, wB.w };
            float* orow = outW + (kg * 8) * N_OUT + ocol;   // ds_add offsets: r*256B
            #pragma unroll
            for (int r = 0; r < 8; ++r) {
                const float v = fast_tanh(acc[r] + bv) * wgt[r];  // v_tanh_f32
                atomicAdd(orow + r * N_OUT, v);                   // ds_add_f32
            }
            // carry (bidx, ocol) to next N-tile
            const int nb   = bidx + 6;
            const bool wr  = nb >= 10;
            bidx = wr ? nb - 10 : nb;
            ocol += wr ? 2 : 1;
        }

        // --- stream the finished 16x64 tile to HBM (float4 stores) ---
        {
            const int p   = lane >> 1;
            const int off = (lane & 1) * 32;
            if (pair0 + p < nPairs) {
                float4* dst        = (float4*)(out + (size_t)(pair0 + p) * N_OUT + off);
                const float4* srcw = (const float4*)(outW + p * N_OUT + off);
                #pragma unroll
                for (int j = 0; j < 8; ++j) dst[j] = srcw[j]; // global_store_b128 x8
            }
        }
    }
}

extern "C" void kernel_launch(void* const* d_in, const int* in_sizes, int n_in,
                              void* d_out, int out_size, void* d_ws, size_t ws_size,
                              hipStream_t stream) {
    (void)n_in; (void)out_size; (void)d_ws; (void)ws_size;
    const int*   ind2  = (const int*)  d_in[0];
    const float* prop  = (const float*)d_in[1];
    const float* basis = (const float*)d_in[2];
    const float* W     = (const float*)d_in[3];
    const float* b     = (const float*)d_in[4];
    float*       out   = (float*)d_out;

    const int nPairs = in_sizes[0] / 2;                       // 500000
    const int nTiles = (nPairs + TILE_M - 1) / TILE_M;        // 31250

    int grid = (nTiles + WAVES - 1) / WAVES;
    if (grid > 512) grid = 512;                               // 1 block per WGP (249KB LDS)

    pilayer_wmma_kernel<<<grid, BLOCK, LDS_TOTAL, stream>>>(
        ind2, prop, basis, W, b, out, nPairs, nTiles);
}
